// LSTM_27496380629854
// MI455X (gfx1250) — compile-verified, matrix operand order checked by reading, not run on hardware
//
#include <hip/hip_runtime.h>
#include <math.h>

typedef __attribute__((ext_vector_type(16))) __bf16 bf16x16;
typedef __attribute__((ext_vector_type(8)))  float  f32x8;

#define BATCH  64
#define SEQ    512
#define IN_DIM 256
#define HID    1024
#define GATES  4096          // 4*HID
#define KTOT   1280          // HID + IN_DIM (fused [h|x] GEMM)
#define NKT    40            // KTOT/32
#define HKT    32            // HID/32
#define XKT    8             // IN_DIM/32
#define NTILES 256           // GATES/16
#define OUTN   1000
#define FC_NT  63            // ceil(1000/16)
#define FC_KT  32            // HID/32

// uint4 strides for A-tile row blocks (16 rows)
#define AH_MSTRIDE 2048      // 16 rows * HID bf16 * 2B / 16B
#define AX_MSTRIDE 262144    // 16 rows * SEQ*IN_DIM bf16 * 2B / 16B

__device__ __forceinline__ unsigned short f2bf(float f) {
    unsigned u = __float_as_uint(f);
    u += 0x7FFFu + ((u >> 16) & 1u);          // round-to-nearest-even
    return (unsigned short)(u >> 16);
}

union Frag { bf16x16 v; uint4 q[2]; };

// ---------------- one-time prep kernels ----------------

__global__ void cvt_x(const float* __restrict__ x, unsigned short* __restrict__ xb, int n) {
    int i = blockIdx.x * blockDim.x + threadIdx.x;
    if (i < n) xb[i] = f2bf(x[i]);
}

// Pack fused weights [W_hh | W_ih] (row g, K=0..1279) into WMMA B-fragment layout:
// Wp[((ntile*NKT + kt)*32 + lane)*8 + v]; lane holds column n = ntile*16 + (lane&15),
// K-halves split by lane>=16 per ISA 7.12.2 (16-bit B layout mirrors A layout).
__global__ void pack_w(const float* __restrict__ Whh, const float* __restrict__ Wih,
                       unsigned int* __restrict__ Wp) {
    int gid = blockIdx.x * blockDim.x + threadIdx.x;
    if (gid >= NTILES * NKT * 32) return;
    int lane = gid & 31;
    int kt   = (gid >> 5) % NKT;
    int nt   = gid / (32 * NKT);
    int n  = nt * 16 + (lane & 15);
    int hi = (lane >> 4) * 8;
    int k0 = kt * 32;
    unsigned int o[8];
    for (int v = 0; v < 8; ++v) {
        int klo = hi + ((v < 4) ? (2 * v) : (16 + 2 * (v - 4)));
        int k = k0 + klo;
        float w0, w1;
        if (k < HID) { w0 = Whh[n * HID + k];            w1 = Whh[n * HID + k + 1]; }
        else         { int kk = k - HID;
                       w0 = Wih[n * IN_DIM + kk];        w1 = Wih[n * IN_DIM + kk + 1]; }
        o[v] = (unsigned)f2bf(w0) | ((unsigned)f2bf(w1) << 16);
    }
    unsigned int* dst = Wp + (size_t)gid * 8;
    for (int v = 0; v < 8; ++v) dst[v] = o[v];
}

__global__ void pack_fc(const float* __restrict__ fcw, unsigned int* __restrict__ Fp) {
    int gid = blockIdx.x * blockDim.x + threadIdx.x;
    if (gid >= FC_NT * FC_KT * 32) return;
    int lane = gid & 31;
    int kt   = (gid >> 5) % FC_KT;
    int nt   = gid / (32 * FC_KT);
    int n  = nt * 16 + (lane & 15);
    int hi = (lane >> 4) * 8;
    int k0 = kt * 32;
    unsigned int o[8];
    for (int v = 0; v < 8; ++v) {
        int klo = hi + ((v < 4) ? (2 * v) : (16 + 2 * (v - 4)));
        int k = k0 + klo;
        float w0 = 0.f, w1 = 0.f;
        if (n < OUTN) { w0 = fcw[n * HID + k]; w1 = fcw[n * HID + k + 1]; }
        o[v] = (unsigned)f2bf(w0) | ((unsigned)f2bf(w1) << 16);
    }
    unsigned int* dst = Fp + (size_t)gid * 8;
    for (int v = 0; v < 8; ++v) dst[v] = o[v];
}

__global__ void init_state(const float* __restrict__ bih, const float* __restrict__ bhh,
                           float* __restrict__ bias4,
                           unsigned short* __restrict__ h0, float* __restrict__ c) {
    int i = blockIdx.x * blockDim.x + threadIdx.x;
    if (i < GATES) bias4[i] = bih[i] + bhh[i];
    if (i < BATCH * HID) { h0[i] = 0; c[i] = 0.f; }
}

// ---------------- per-timestep kernel ----------------
// grid.x = 64 (one block per 16 hidden columns), block = 128 threads = 4 waves.
// wave = gate; each wave computes all 4 M-tiles (full batch) for its gate.
// Single fused 40-iteration K loop (A-source select is compile-time under full
// unroll). B stream (L2-latency) is 2-deep double-buffered: slot kt&1 is consumed
// by iteration kt and reloaded for kt+2, giving each B fetch ~4 WMMAs of slack.
__global__ __launch_bounds__(128) void lstm_step(
    const unsigned short* __restrict__ h_in,
    unsigned short* __restrict__ h_out,
    float* __restrict__ c,
    const unsigned short* __restrict__ xb,
    const unsigned int* __restrict__ Wp,
    const float* __restrict__ bias4,
    int t)
{
    __shared__ float g4[4][BATCH][17];     // pad 17 to break LDS bank conflicts

    int tid  = threadIdx.x;
    int lane = tid & 31;
    int gate = tid >> 5;                   // wave index = gate 0..3
    int bx   = blockIdx.x;                 // hidden-column tile 0..63
    int ntile = gate * 64 + bx;            // column tile in [4H]
    int r   = lane & 15;
    int hi8 = (lane >> 4) * 8;

    f32x8 acc[4];
#pragma unroll
    for (int m = 0; m < 4; ++m) acc[m] = (f32x8){0,0,0,0,0,0,0,0};

    // loop-invariant base pointers (16B aligned)
    const uint4* wq = (const uint4*)(Wp + ((size_t)(ntile * NKT) * 32 + lane) * 8);
    const uint4* ah = (const uint4*)(h_in + (size_t)r * HID + hi8);
    const uint4* ax = (const uint4*)(xb + (size_t)r * (SEQ * IN_DIM)
                                        + (size_t)t * IN_DIM + hi8);

    // B double-buffer prologue: kt=0 and kt=1
    Frag Bb[2];
    Bb[0].q[0] = wq[0];   Bb[0].q[1] = wq[1];
    Bb[1].q[0] = wq[64];  Bb[1].q[1] = wq[65];

#pragma unroll
    for (int kt = 0; kt < NKT; ++kt) {
        // A fragments (source select is compile-time constant under unroll)
        Frag A[4];
        if (kt < HKT) {
#pragma unroll
            for (int m = 0; m < 4; ++m) {
                A[m].q[0] = ah[kt * 4 + m * AH_MSTRIDE];
                A[m].q[1] = ah[kt * 4 + 2 + m * AH_MSTRIDE];
            }
        } else {
#pragma unroll
            for (int m = 0; m < 4; ++m) {
                A[m].q[0] = ax[(kt - HKT) * 4 + m * AX_MSTRIDE];
                A[m].q[1] = ax[(kt - HKT) * 4 + 2 + m * AX_MSTRIDE];
            }
        }
#pragma unroll
        for (int m = 0; m < 4; ++m)
            acc[m] = __builtin_amdgcn_wmma_f32_16x16x32_bf16(false, A[m].v, false, Bb[kt & 1].v,
                                                             (short)0, acc[m], false, false);
        if (kt + 2 < NKT) {                // refill consumed slot for iteration kt+2
            Bb[kt & 1].q[0] = wq[(kt + 2) * 64];
            Bb[kt & 1].q[1] = wq[(kt + 2) * 64 + 1];
        }
    }

    // C/D layout: VGPR j -> row (lane>=16 ? j+8 : j), col lane&15
    int rb = (lane >> 4) * 8;
#pragma unroll
    for (int m = 0; m < 4; ++m)
#pragma unroll
        for (int j = 0; j < 8; ++j)
            g4[gate][m * 16 + rb + j][r] = acc[m][j];
    __syncthreads();

    // pointwise gate math for this block's 16 columns x 64 batch rows
    for (int idx = tid; idx < BATCH * 16; idx += 128) {
        int b = idx >> 4, col = idx & 15;
        int n = bx * 16 + col;
        float iv = g4[0][b][col] + bias4[n];
        float fv = g4[1][b][col] + bias4[HID + n];
        float gv = g4[2][b][col] + bias4[2 * HID + n];
        float ov = g4[3][b][col] + bias4[3 * HID + n];
        iv = 1.f / (1.f + __expf(-iv));
        fv = 1.f / (1.f + __expf(-fv));
        gv = tanhf(gv);
        ov = 1.f / (1.f + __expf(-ov));
        float cn = fv * c[b * HID + n] + iv * gv;
        c[b * HID + n] = cn;
        h_out[b * HID + n] = f2bf(ov * tanhf(cn));
    }
}

// ---------------- FC head: out = h @ fc_w^T + fc_b ----------------
// one wave per block, 4 M-tiles per wave, B double-buffered.
__global__ __launch_bounds__(32) void fc_kernel(
    const unsigned short* __restrict__ h,
    const unsigned int* __restrict__ Fp,
    const float* __restrict__ fcb,
    float* __restrict__ out)
{
    int lane = threadIdx.x;
    int nt = blockIdx.x;                    // 0..62
    int r = lane & 15, hi8 = (lane >> 4) * 8;

    f32x8 acc[4];
#pragma unroll
    for (int m = 0; m < 4; ++m) acc[m] = (f32x8){0,0,0,0,0,0,0,0};

    const uint4* fq = (const uint4*)(Fp + ((size_t)(nt * FC_KT) * 32 + lane) * 8);
    const uint4* aq = (const uint4*)(h + (size_t)r * HID + hi8);

    Frag Bb[2];
    Bb[0].q[0] = fq[0];   Bb[0].q[1] = fq[1];
    Bb[1].q[0] = fq[64];  Bb[1].q[1] = fq[65];

#pragma unroll
    for (int kt = 0; kt < FC_KT; ++kt) {
        Frag A[4];
#pragma unroll
        for (int m = 0; m < 4; ++m) {
            A[m].q[0] = aq[kt * 4 + m * AH_MSTRIDE];
            A[m].q[1] = aq[kt * 4 + 2 + m * AH_MSTRIDE];
        }
#pragma unroll
        for (int m = 0; m < 4; ++m)
            acc[m] = __builtin_amdgcn_wmma_f32_16x16x32_bf16(false, A[m].v, false, Bb[kt & 1].v,
                                                             (short)0, acc[m], false, false);
        if (kt + 2 < FC_KT) {
            Bb[kt & 1].q[0] = fq[(kt + 2) * 64];
            Bb[kt & 1].q[1] = fq[(kt + 2) * 64 + 1];
        }
    }

    int n = nt * 16 + r;
    int rb = (lane >> 4) * 8;
    if (n < OUTN) {
#pragma unroll
        for (int m = 0; m < 4; ++m)
#pragma unroll
            for (int j = 0; j < 8; ++j)
                out[(size_t)(m * 16 + rb + j) * OUTN + n] = acc[m][j] + fcb[n];
    }
}

// ---------------- host ----------------

extern "C" void kernel_launch(void* const* d_in, const int* in_sizes, int n_in,
                              void* d_out, int out_size, void* d_ws, size_t ws_size,
                              hipStream_t stream) {
    (void)in_sizes; (void)n_in; (void)out_size; (void)ws_size;

    const float* x   = (const float*)d_in[0];
    const float* Wih = (const float*)d_in[1];
    const float* Whh = (const float*)d_in[2];
    const float* bih = (const float*)d_in[3];
    const float* bhh = (const float*)d_in[4];
    const float* fcw = (const float*)d_in[5];
    const float* fcb = (const float*)d_in[6];
    float* out = (float*)d_out;

    char* ws = (char*)d_ws;
    size_t off = 0;
    auto wsalloc = [&](size_t bytes) -> void* {
        void* p = ws + off;
        off = (off + bytes + 255) & ~(size_t)255;
        return p;
    };

    unsigned short* xb = (unsigned short*)wsalloc((size_t)BATCH * SEQ * IN_DIM * 2);  // 16 MB
    unsigned int*   Wp = (unsigned int*)  wsalloc((size_t)NTILES * NKT * 32 * 8 * 4); // 10.5 MB (fits L2)
    unsigned int*   Fp = (unsigned int*)  wsalloc((size_t)FC_NT * FC_KT * 32 * 8 * 4);
    float*       bias4 = (float*)         wsalloc((size_t)GATES * 4);
    unsigned short* h0 = (unsigned short*)wsalloc((size_t)BATCH * HID * 2);
    unsigned short* h1 = (unsigned short*)wsalloc((size_t)BATCH * HID * 2);
    float*           c = (float*)         wsalloc((size_t)BATCH * HID * 4);

    {
        int n = BATCH * SEQ * IN_DIM;
        cvt_x<<<(n + 255) / 256, 256, 0, stream>>>(x, xb, n);
    }
    pack_w<<<(NTILES * NKT * 32 + 255) / 256, 256, 0, stream>>>(Whh, Wih, Wp);
    pack_fc<<<(FC_NT * FC_KT * 32 + 255) / 256, 256, 0, stream>>>(fcw, Fp);
    init_state<<<(BATCH * HID + 255) / 256, 256, 0, stream>>>(bih, bhh, bias4, h0, c);

    // sequential recurrence, ping-pong h buffers (t even: h0->h1, t odd: h1->h0)
    for (int t = 0; t < SEQ; ++t) {
        const unsigned short* hin  = (t & 1) ? h1 : h0;
        unsigned short*       hout = (t & 1) ? h0 : h1;
        lstm_step<<<64, 128, 0, stream>>>(hin, hout, c, xb, Wp, bias4, t);
    }
    // SEQ=512 even -> final hidden lands in h0
    fc_kernel<<<FC_NT, 32, 0, stream>>>(h0, Fp, fcb, out);
}
